// HypergraphNodeAttentionBlock_58669253264112
// MI455X (gfx1250) — matrix operand order repaired and checked by reading, not compiled, not run on hardware
//
#include <hip/hip_runtime.h>

// ---------------- problem constants (fixed by the harness) ----------------
#define NN      100000
#define EE      400000
#define TT      400000
#define GG      4
#define DN      128
#define DE      128
#define DH      64
#define DG      16
#define AK      8
#define HH      8
#define DIN     216
#define KPAD    224          // DIN padded to 7 * 32
#define H1      256
#define LN_EPS  1e-3f

typedef __attribute__((ext_vector_type(16))) __bf16 v16bf;
typedef __attribute__((ext_vector_type(8)))  __bf16 v8bf;
typedef __attribute__((ext_vector_type(8)))  float  v8f;

// ---------------- workspace layout (bytes) ----------------
#define WS_WQ2    ((size_t)0)                       // 128*8 f32
#define WS_WK2    ((size_t)4096)                    // 128*8 f32
#define WS_BQ2    ((size_t)8192)                    // 8 f32
#define WS_BK2    ((size_t)8448)                    // 8 f32
#define WS_W1T    ((size_t)16384)                   // 256*224 bf16
#define WS_W2T    ((size_t)131072)                  // 128*256 bf16
#define WS_QN     ((size_t)196608)                  // N*8 f32
#define WS_KV     ((size_t)3396608)                 // E*8 f32
#define WS_M      ((size_t)16196608)                // N*8 u32 (encoded max)
#define WS_DEN    ((size_t)19396608)                // N*8 f32
#define WS_NUM    ((size_t)22596608)                // N*8 f32
#define WS_HYP    ((size_t)25796608)                // N*64 f32

__device__ __forceinline__ unsigned fenc(float f) {
    unsigned u = __float_as_uint(f);
    return (u >> 31) ? ~u : (u | 0x80000000u);
}
__device__ __forceinline__ float fdec(unsigned k) {
    return (k & 0x80000000u) ? __uint_as_float(k & 0x7FFFFFFFu)
                             : __uint_as_float(~k);
}
__device__ __forceinline__ v16bf cat8(v8bf lo, v8bf hi) {
    return __builtin_shufflevector(lo, hi, 0,1,2,3,4,5,6,7,8,9,10,11,12,13,14,15);
}
__device__ __forceinline__ v8f wmma_bf16(v16bf a, v16bf b, v8f c) {
    return __builtin_amdgcn_wmma_f32_16x16x32_bf16(false, a, false, b, (short)0, c, false, false);
}

// ---------------- K0: fold Wmh into Wq/Wk; bf16-transpose W1/W2 ----------------
__global__ void k_prep(const float* Wq, const float* bq, const float* Wk, const float* bk,
                       const float* Wmh, const float* bmh, const float* W1, const float* W2,
                       char* ws) {
    float*  Wq2 = (float*)(ws + WS_WQ2);
    float*  Wk2 = (float*)(ws + WS_WK2);
    float*  bq2 = (float*)(ws + WS_BQ2);
    float*  bk2 = (float*)(ws + WS_BK2);
    __bf16* W1t = (__bf16*)(ws + WS_W1T);
    __bf16* W2t = (__bf16*)(ws + WS_W2T);

    const int gid = blockIdx.x * blockDim.x + threadIdx.x;
    const int gsz = gridDim.x * blockDim.x;

    for (int i = gid; i < DN * HH; i += gsz) {          // Wq2/Wk2 = W @ Wmh
        int c = i / HH, h = i % HH;
        float aq = 0.f, ak = 0.f;
        for (int a = 0; a < AK; ++a) {
            aq += Wq[c * AK + a] * Wmh[a * HH + h];
            ak += Wk[c * AK + a] * Wmh[a * HH + h];
        }
        Wq2[i] = aq; Wk2[i] = ak;
    }
    for (int h = gid; h < HH; h += gsz) {               // folded biases
        float aq = bmh[h], ak = bmh[h];
        for (int a = 0; a < AK; ++a) {
            aq += bq[a] * Wmh[a * HH + h];
            ak += bk[a] * Wmh[a * HH + h];
        }
        bq2[h] = aq; bk2[h] = ak;
    }
    for (int i = gid; i < H1 * KPAD; i += gsz) {        // W1t[n][k] = W1[k][n], K padded
        int n = i / KPAD, k = i % KPAD;
        W1t[i] = (k < DIN) ? (__bf16)W1[k * H1 + n] : (__bf16)0.f;
    }
    for (int i = gid; i < DN * H1; i += gsz) {          // W2t[n][k] = W2[k][n]
        int n = i / H1, k = i % H1;
        W2t[i] = (__bf16)W2[k * DN + n];
    }
}

// ---------------- Kinit: clear accumulators ----------------
__global__ void k_init(char* ws) {
    unsigned* m   = (unsigned*)(ws + WS_M);
    float*    den = (float*)(ws + WS_DEN);
    float*    num = (float*)(ws + WS_NUM);
    float*    hyp = (float*)(ws + WS_HYP);
    const long i = (long)blockIdx.x * blockDim.x + threadIdx.x;
    if (i < (long)NN * HH) { m[i] = fenc(-__builtin_inff()); den[i] = 0.f; num[i] = 0.f; }
    if (i < (long)NN * DH) hyp[i] = 0.f;
}

// ---------------- Kq: q = nodes @ Wq2 + bq2 ----------------
__global__ __launch_bounds__(256) void k_qnodes(const float* nodes, char* ws) {
    __shared__ float sW[DN * HH];
    __shared__ float sB[HH];
    const float* Wq2 = (const float*)(ws + WS_WQ2);
    const float* bq2 = (const float*)(ws + WS_BQ2);
    for (int i = threadIdx.x; i < DN * HH; i += blockDim.x) sW[i] = Wq2[i];
    if (threadIdx.x < HH) sB[threadIdx.x] = bq2[threadIdx.x];
    __syncthreads();

    const int node = blockIdx.x * blockDim.x + threadIdx.x;
    if (node >= NN) return;
    float acc[HH];
    #pragma unroll
    for (int h = 0; h < HH; ++h) acc[h] = sB[h];
    const float4* row = (const float4*)(nodes + (long)node * DN);
    #pragma unroll 4
    for (int i = 0; i < DN / 4; ++i) {
        float4 v = row[i];
        const float* vp = &v.x;
        #pragma unroll
        for (int j = 0; j < 4; ++j) {
            float x = vp[j]; int c = i * 4 + j;
            #pragma unroll
            for (int h = 0; h < HH; ++h) acc[h] += x * sW[c * HH + h];
        }
    }
    float* qn = (float*)(ws + WS_QN);
    #pragma unroll
    for (int h = 0; h < HH; ++h) qn[(long)node * HH + h] = acc[h];
}

// ---------------- Kedge pass 1: kv + segment-max (encoded atomicMax) ----------------
__global__ __launch_bounds__(256) void k_edge_max(const float* edges, const int* eidx, char* ws) {
    __shared__ float sW[DE * HH];
    __shared__ float sB[HH];
    const float* Wk2 = (const float*)(ws + WS_WK2);
    const float* bk2 = (const float*)(ws + WS_BK2);
    for (int i = threadIdx.x; i < DE * HH; i += blockDim.x) sW[i] = Wk2[i];
    if (threadIdx.x < HH) sB[threadIdx.x] = bk2[threadIdx.x];
    __syncthreads();

    const int e = blockIdx.x * blockDim.x + threadIdx.x;
    if (e >= EE) return;
    float kvv[HH];
    #pragma unroll
    for (int h = 0; h < HH; ++h) kvv[h] = sB[h];
    const float4* row = (const float4*)(edges + (long)e * DE);
    #pragma unroll 4
    for (int i = 0; i < DE / 4; ++i) {
        float4 v = row[i];
        const float* vp = &v.x;
        #pragma unroll
        for (int j = 0; j < 4; ++j) {
            float x = vp[j]; int c = i * 4 + j;
            #pragma unroll
            for (int h = 0; h < HH; ++h) kvv[h] += x * sW[c * HH + h];
        }
    }
    float* kv = (float*)(ws + WS_KV);
    #pragma unroll
    for (int h = 0; h < HH; ++h) kv[(long)e * HH + h] = kvv[h];

    const float* qn = (const float*)(ws + WS_QN);
    unsigned*    m  = (unsigned*)(ws + WS_M);
    #pragma unroll
    for (int s = 0; s < 2; ++s) {
        int node = eidx[e * 2 + s];
        #pragma unroll
        for (int h = 0; h < HH; ++h) {
            float logit = qn[(long)node * HH + h] * kvv[h];
            atomicMax(&m[(long)node * HH + h], fenc(logit));
        }
    }
}

// ---------------- Kedge pass 2: softmax numer/denom scatter ----------------
__global__ __launch_bounds__(256) void k_edge_sum(const int* eidx, char* ws) {
    const int e = blockIdx.x * blockDim.x + threadIdx.x;
    if (e >= EE) return;
    const float*    kv  = (const float*)(ws + WS_KV);
    const float*    qn  = (const float*)(ws + WS_QN);
    const unsigned* m   = (const unsigned*)(ws + WS_M);
    float*          den = (float*)(ws + WS_DEN);
    float*          num = (float*)(ws + WS_NUM);
    float kvv[HH];
    #pragma unroll
    for (int h = 0; h < HH; ++h) kvv[h] = kv[(long)e * HH + h];
    #pragma unroll
    for (int s = 0; s < 2; ++s) {
        int node = eidx[e * 2 + s];
        #pragma unroll
        for (int h = 0; h < HH; ++h) {
            float logit = qn[(long)node * HH + h] * kvv[h];
            float w = __expf(logit - fdec(m[(long)node * HH + h]));
            atomicAdd(&den[(long)node * HH + h], w);
            atomicAdd(&num[(long)node * HH + h], w * kvv[h]);
        }
    }
}

// ---------------- Khyper: scatter-sum hyperedges onto 3 incident nodes ----------------
__global__ __launch_bounds__(256) void k_hyper(const float* hyperedges, const int* hidx, char* ws) {
    const int t = blockIdx.x * blockDim.x + threadIdx.x;
    if (t >= TT) return;
    float* hyp = (float*)(ws + WS_HYP);
    float v[DH];
    const float4* row = (const float4*)(hyperedges + (long)t * DH);
    #pragma unroll
    for (int i = 0; i < DH / 4; ++i) {
        float4 x = row[i];
        v[i*4+0] = x.x; v[i*4+1] = x.y; v[i*4+2] = x.z; v[i*4+3] = x.w;
    }
    #pragma unroll
    for (int s = 0; s < 3; ++s) {
        long base = (long)hidx[t * 3 + s] * DH;
        #pragma unroll 8
        for (int c = 0; c < DH; ++c) atomicAdd(&hyp[base + c], v[c]);
    }
}

// ---------------- K5: fused  x -> relu(xW1+b1) -> relu(.W2+b2) -> LayerNorm ----------------
// 4 waves / block, 1 wave = one 16-node tile. bf16 WMMA, fp32 accumulate.
__global__ __launch_bounds__(128) void k_mlp(const float* nodes, const float* globals_,
                                             const float* b1, const float* b2,
                                             const float* gamma, const float* beta,
                                             const int* n_node, const char* ws, float* out) {
    __shared__ __align__(16) __bf16 sX[4][16][KPAD];   // x tile per wave
    __shared__ __align__(16) __bf16 sH[4][16][H1];     // h1 tile per wave

    const int wid  = threadIdx.x >> 5;
    const int lane = threadIdx.x & 31;
    const int tile = blockIdx.x * 4 + wid;
    if (tile >= NN / 16) return;                       // wave-uniform exit
    const int nodeBase = tile * 16;

    const __bf16* W1t = (const __bf16*)(ws + WS_W1T);
    const __bf16* W2t = (const __bf16*)(ws + WS_W2T);
    const float*  den = (const float*)(ws + WS_DEN);
    const float*  num = (const float*)(ws + WS_NUM);
    const float*  hyp = (const float*)(ws + WS_HYP);
    const int perGraph = n_node[0];

    // ---- stage x tile (bf16) into LDS: [nodes | globals | att | hyper | pad] ----
    for (int i = lane; i < 16 * KPAD; i += 32) {
        int r = i / KPAD, c = i % KPAD;
        int node = nodeBase + r;
        float v;
        if (c < DN) {
            v = nodes[(long)node * DN + c];
        } else if (c < DN + DG) {
            v = globals_[(node / perGraph) * DG + (c - DN)];
        } else if (c < DN + DG + HH) {
            int h = c - DN - DG;
            float d = den[(long)node * HH + h];
            v = num[(long)node * HH + h] / ((d == 0.f) ? 1.f : d);
        } else if (c < DIN) {
            v = hyp[(long)node * DH + (c - DN - DG - HH)];
        } else {
            v = 0.f;
        }
        sX[wid][r][c] = (__bf16)v;
    }

    const int jcol  = lane & 15;        // column within 16-wide N tile / row for A
    const int khalf = lane >> 4;        // 0: lanes 0-15, 1: lanes 16-31

    // ---- GEMM1: 16x256 = 16x224 @ 224x256, relu -> sH ----
    for (int t = 0; t < 16; ++t) {
        const int nCol = t * 16 + jcol;
        float bv = b1[nCol];
        v8f c = {bv, bv, bv, bv, bv, bv, bv, bv};
        #pragma unroll
        for (int kb = 0; kb < KPAD / 32; ++kb) {
            const int ka = kb * 32 + khalf * 8;                     // A: K {0..7,16..23}/{8..15,24..31}
            v8bf a0 = *(const v8bf*)&sX[wid][jcol][ka];
            v8bf a1 = *(const v8bf*)&sX[wid][jcol][ka + 16];
            const int kbB = kb * 32 + khalf * 8;                    // B: K {0..7,16..23}/{8..15,24..31}
            const v8bf* pb = (const v8bf*)(W1t + (long)nCol * KPAD + kbB);
            c = wmma_bf16(cat8(a0, a1), cat8(pb[0], pb[2]), c);
        }
        #pragma unroll
        for (int r = 0; r < 8; ++r)
            sH[wid][(khalf << 3) + r][nCol] = (__bf16)fmaxf(c[r], 0.f);
    }

    // ---- GEMM2: 16x128 = 16x256 @ 256x128, relu ----
    v8f acc[8];
    #pragma unroll
    for (int t2 = 0; t2 < 8; ++t2) {
        float bv = b2[t2 * 16 + jcol];
        v8f c = {bv, bv, bv, bv, bv, bv, bv, bv};
        #pragma unroll
        for (int kb = 0; kb < H1 / 32; ++kb) {
            const int ka = kb * 32 + khalf * 8;
            v8bf a0 = *(const v8bf*)&sH[wid][jcol][ka];
            v8bf a1 = *(const v8bf*)&sH[wid][jcol][ka + 16];
            const int kbB = kb * 32 + khalf * 8;
            const v8bf* pb = (const v8bf*)(W2t + (long)(t2 * 16 + jcol) * H1 + kbB);
            c = wmma_bf16(cat8(a0, a1), cat8(pb[0], pb[2]), c);
        }
        #pragma unroll
        for (int r = 0; r < 8; ++r) c[r] = fmaxf(c[r], 0.f);
        acc[t2] = c;
    }

    // ---- LayerNorm per row (128 values spread over 16 lanes x 8 n-tiles) ----
    #pragma unroll
    for (int r = 0; r < 8; ++r) {
        float s = 0.f, ss = 0.f;
        #pragma unroll
        for (int t2 = 0; t2 < 8; ++t2) { float v = acc[t2][r]; s += v; ss += v * v; }
        #pragma unroll
        for (int msk = 1; msk < 16; msk <<= 1) {
            s  += __shfl_xor(s,  msk, 16);
            ss += __shfl_xor(ss, msk, 16);
        }
        const float mean = s * (1.f / DN);
        const float var  = ss * (1.f / DN) - mean * mean;
        const float inv  = rsqrtf(var + LN_EPS);
        const int node   = nodeBase + (khalf << 3) + r;
        #pragma unroll
        for (int t2 = 0; t2 < 8; ++t2) {
            int nCol = t2 * 16 + jcol;
            out[(long)node * DN + nCol] =
                gamma[nCol] * (acc[t2][r] - mean) * inv + beta[nCol];
        }
    }
}

// ---------------- launcher ----------------
extern "C" void kernel_launch(void* const* d_in, const int* in_sizes, int n_in,
                              void* d_out, int out_size, void* d_ws, size_t ws_size,
                              hipStream_t stream) {
    const float* nodes      = (const float*)d_in[0];
    const float* edges      = (const float*)d_in[1];
    const float* hyperedges = (const float*)d_in[2];
    const float* globals_   = (const float*)d_in[3];
    const float* Wq         = (const float*)d_in[4];
    const float* bq         = (const float*)d_in[5];
    const float* Wk         = (const float*)d_in[6];
    const float* bk         = (const float*)d_in[7];
    const float* Wmh        = (const float*)d_in[8];
    const float* bmh        = (const float*)d_in[9];
    const float* W1         = (const float*)d_in[10];
    const float* b1         = (const float*)d_in[11];
    const float* W2         = (const float*)d_in[12];
    const float* b2         = (const float*)d_in[13];
    const float* gamma      = (const float*)d_in[14];
    const float* beta       = (const float*)d_in[15];
    const int*   eidx       = (const int*)d_in[16];
    const int*   hidx       = (const int*)d_in[17];
    const int*   n_node     = (const int*)d_in[18];
    char* ws  = (char*)d_ws;
    float* out = (float*)d_out;

    k_prep<<<64, 256, 0, stream>>>(Wq, bq, Wk, bk, Wmh, bmh, W1, W2, ws);
    k_init<<<((long)NN * DH + 255) / 256, 256, 0, stream>>>(ws);
    k_qnodes<<<(NN + 255) / 256, 256, 0, stream>>>(nodes, ws);
    k_edge_max<<<(EE + 255) / 256, 256, 0, stream>>>(edges, eidx, ws);
    k_edge_sum<<<(EE + 255) / 256, 256, 0, stream>>>(eidx, ws);
    k_hyper<<<(TT + 255) / 256, 256, 0, stream>>>(hyperedges, hidx, ws);
    k_mlp<<<(NN / 16 + 3) / 4, 128, 0, stream>>>(nodes, globals_, b1, b2,
                                                 gamma, beta, n_node, ws, out);
}
